// ProductQuantizerV3_53180285059806
// MI455X (gfx1250) — matrix operand order repaired and checked by reading, not compiled, not run on hardware
//
#include <hip/hip_runtime.h>

// ---------------------------------------------------------------------------
// Problem constants (reference: H=8, K=2048, D=64, HID=1024, B=8192)
// ---------------------------------------------------------------------------
#define Bb  8192
#define Hh  8
#define Kk  2048
#define Dd  64
#define HIDN 1024

typedef __attribute__((ext_vector_type(16))) __bf16        v16bf;
typedef __attribute__((ext_vector_type(8)))  float         v8f;

#define DEV __device__ __forceinline__

DEV unsigned short f2bf(float x) {
  unsigned u = __float_as_uint(x);
  unsigned r = (u + 0x7FFFu + ((u >> 16) & 1u)) >> 16;  // round-to-nearest-even
  return (unsigned short)r;
}

struct U8x { uint4 a, b; };

// Load 16 bf16 fragment elements: first 8 at p[0..7], next 8 at p[gap..gap+7].
// gap=16 -> A-fragment pattern (K chunks {0..7} and {16..23} per ISA layout)
// gap=8  -> 16 contiguous values (not used: B-fragments also load with gap=16, same k order as A)
DEV v16bf ld_bf16x16(const unsigned short* p, int gap) {
  U8x u;
  u.a = *(const uint4*)p;
  u.b = *(const uint4*)(p + gap);
  return __builtin_bit_cast(v16bf, u);
}

DEV v8f wmma_bf16(v16bf a, v16bf b, v8f c) {
  // D = A(16x32 bf16) * B(32x16 bf16) + C(16x16 f32)
  return __builtin_amdgcn_wmma_f32_16x16x32_bf16(false, a, false, b,
                                                 (short)0, c, false, false);
}

DEV v8f zero8() { v8f z = {0.f,0.f,0.f,0.f,0.f,0.f,0.f,0.f}; return z; }

// ---------------------------------------------------------------------------
// k_prep: zero accumulators; convert z + weights + codebooks to bf16 ONCE
// (no fp32->bf16 conversion survives in any GEMM inner loop)
//   zbf  [b][i]     straight convert of z
//   w1t  [h][o][k]  (o 0..127, k 0..1023)   transposed for B-fragments
//   w2t  [h][d][k]  (d 0..63,  k 0..127)
//   dwt  [h][o][d]  (o 0..127, d 0..63)
//   cbbf [h][k][d]  straight convert
// ---------------------------------------------------------------------------
__global__ void k_prep(const float* __restrict__ z, const float* __restrict__ w1,
                       const float* __restrict__ w2, const float* __restrict__ decw,
                       const float* __restrict__ cb,
                       unsigned short* __restrict__ zbf,
                       unsigned short* __restrict__ w1t, unsigned short* __restrict__ w2t,
                       unsigned short* __restrict__ dwt, unsigned short* __restrict__ cbbf,
                       float* __restrict__ co, float* __restrict__ avgp,
                       float* __restrict__ pcnt, float* __restrict__ scal) {
  int t = blockIdx.x * blockDim.x + threadIdx.x;
  int str = gridDim.x * blockDim.x;
  for (int i = t; i < Kk*Kk; i += str) co[i] = 0.f;
  for (int i = t; i < Hh*Kk; i += str) { avgp[i] = 0.f; pcnt[i] = 0.f; }
  for (int i = t; i < 8; i += str) scal[i] = 0.f;
  for (int i = t; i < Bb*HIDN; i += str) zbf[i] = f2bf(z[i]);
  for (int i = t; i < Hh*128*HIDN; i += str) {
    int h = i >> 17, rem = i & 131071, o = rem >> 10, k = rem & 1023;
    w1t[i] = f2bf(w1[((size_t)h*HIDN + k)*128 + o]);
  }
  for (int i = t; i < Hh*64*128; i += str) {
    int h = i >> 13, rem = i & 8191, dd = rem >> 7, k = rem & 127;
    w2t[i] = f2bf(w2[((size_t)h*128 + k)*64 + dd]);
  }
  for (int i = t; i < Hh*128*64; i += str) {
    int h = i >> 13, rem = i & 8191, o = rem >> 6, dd = rem & 63;
    dwt[i] = f2bf(decw[((size_t)h*64 + dd)*128 + o]);
  }
  for (int i = t; i < Hh*Kk*64; i += str) cbbf[i] = f2bf(cb[i]);
}

__global__ void k_csq(const float* __restrict__ cb, float* __restrict__ csq) {
  int i = blockIdx.x * blockDim.x + threadIdx.x;
  if (i < Hh*Kk) {
    const float* p = cb + (size_t)i * 64;
    float s = 0.f;
#pragma unroll 8
    for (int d = 0; d < 64; ++d) s += p[d]*p[d];
    csq[i] = s;
  }
}

// ---------------------------------------------------------------------------
// k1: h1 = gelu(z @ W1cat + b1)   M=8192 N=1024 K=1024, bf16 WMMA, f32 accum
// Block = 256 thr (8 waves), tile 128x128; wave = 32x64 (2x4 fragments).
// Inner loop: pure b128 loads + 8 WMMA. h1 written directly in bf16 (only
// ever consumed as a WMMA A-operand by k2).
// ---------------------------------------------------------------------------
__global__ __launch_bounds__(256) void k1_gemm1_gelu(
    const unsigned short* __restrict__ zbf, const unsigned short* __restrict__ w1t,
    const float* __restrict__ b1, unsigned short* __restrict__ h1bf) {
  const int lane = threadIdx.x & 31, wave = threadIdx.x >> 5;
  const int r = lane & 15, sel = lane >> 4;
  const int m0 = blockIdx.y * 128 + (wave >> 1) * 32;
  const int n0 = blockIdx.x * 128 + (wave & 1) * 64;

  v8f acc[2][4];
#pragma unroll
  for (int mi = 0; mi < 2; ++mi)
#pragma unroll
    for (int ni = 0; ni < 4; ++ni) acc[mi][ni] = zero8();

  const unsigned short* aBase = zbf + (size_t)(m0 + r) * HIDN + sel * 8;
  const unsigned short* bBase = w1t + (size_t)(n0 + r) * HIDN + sel * 8;

  for (int k0 = 0; k0 < HIDN; k0 += 32) {
    v16bf a[2], b[4];
#pragma unroll
    for (int mi = 0; mi < 2; ++mi) a[mi] = ld_bf16x16(aBase + mi*16*HIDN + k0, 16);
#pragma unroll
    for (int ni = 0; ni < 4; ++ni) b[ni] = ld_bf16x16(bBase + ni*16*HIDN + k0, 16);
#pragma unroll
    for (int mi = 0; mi < 2; ++mi)
#pragma unroll
      for (int ni = 0; ni < 4; ++ni)
        acc[mi][ni] = wmma_bf16(a[mi], b[ni], acc[mi][ni]);
  }
#pragma unroll
  for (int mi = 0; mi < 2; ++mi)
#pragma unroll
    for (int ni = 0; ni < 4; ++ni) {
      int col = n0 + ni*16 + r;
      float bias = b1[col];                       // b1 flat (H,128) == col
#pragma unroll
      for (int rr = 0; rr < 8; ++rr) {
        int row = m0 + mi*16 + sel*8 + rr;
        float x = acc[mi][ni][rr] + bias;
        float g = 0.5f * x * (1.0f + erff(x * 0.70710678118f));  // exact gelu
        h1bf[(size_t)row * HIDN + col] = f2bf(g);
      }
    }
}

// ---------------------------------------------------------------------------
// k2: zh = h1[:,h] @ w2[h] + b2; LayerNorm -> z_heads (f32 to d_out + bf16) +
// per-row |z|^2.  Grid (H, B/128). Wave = 16 rows x 64 cols, K=128 in 4 steps.
// ---------------------------------------------------------------------------
__global__ __launch_bounds__(256) void k2_gemm2_ln(
    const unsigned short* __restrict__ h1bf, const unsigned short* __restrict__ w2t,
    const float* __restrict__ b2, const float* __restrict__ lng,
    const float* __restrict__ lnb,
    unsigned short* __restrict__ zhbf, float* __restrict__ zsq,
    float* __restrict__ out_zh) {
  __shared__ float t[128][65];
  const int lane = threadIdx.x & 31, wave = threadIdx.x >> 5;
  const int r = lane & 15, sel = lane >> 4;
  const int h = blockIdx.x;
  const int m0 = blockIdx.y * 128;
  const int mrow = m0 + wave * 16;

  v8f acc[4];
#pragma unroll
  for (int ni = 0; ni < 4; ++ni) acc[ni] = zero8();

  const unsigned short* aBase = h1bf + (size_t)(mrow + r) * HIDN + h*128 + sel*8;
  const unsigned short* bBase = w2t + (size_t)h * (64*128) + r*128 + sel*8;
#pragma unroll
  for (int k0 = 0; k0 < 128; k0 += 32) {
    v16bf a = ld_bf16x16(aBase + k0, 16);
#pragma unroll
    for (int ni = 0; ni < 4; ++ni)
      acc[ni] = wmma_bf16(a, ld_bf16x16(bBase + ni*16*128 + k0, 16), acc[ni]);
  }
#pragma unroll
  for (int ni = 0; ni < 4; ++ni)
#pragma unroll
    for (int rr = 0; rr < 8; ++rr)
      t[wave*16 + sel*8 + rr][ni*16 + r] = acc[ni][rr];
  __syncthreads();

  if (threadIdx.x < 128) {
    int lrow = threadIdx.x;
    int row = m0 + lrow;
    float mu = 0.f, s2 = 0.f;
#pragma unroll 8
    for (int d = 0; d < 64; ++d) {
      float x = t[lrow][d] + b2[h*64 + d];
      mu += x; s2 += x*x;
    }
    mu *= (1.0f/64.0f); s2 *= (1.0f/64.0f);
    float rstd = rsqrtf(s2 - mu*mu + 1e-5f);
    float q = 0.f;
    size_t base = ((size_t)row * Hh + h) * 64;
#pragma unroll 8
    for (int d = 0; d < 64; ++d) {
      float x = t[lrow][d] + b2[h*64 + d];
      float y = (x - mu) * rstd * lng[h*64 + d] + lnb[h*64 + d];
      zhbf[base + d]   = f2bf(y);
      out_zh[base + d] = y;        // output #7: sg(z_heads)
      q += y*y;
    }
    zsq[(size_t)row * Hh + h] = q;
  }
}

// ---------------------------------------------------------------------------
// k3a: dists = |z|^2 + |c|^2 - 2 z.c via WMMA; argmin + online logsumexp(-d)
// Grid (B/128, H); wave owns 16 rows, loops K=2048 in 16-entry fragments.
// ---------------------------------------------------------------------------
__global__ __launch_bounds__(256) void k3a_dist(
    const unsigned short* __restrict__ zbf, const unsigned short* __restrict__ cbbf,
    const float* __restrict__ csq, const float* __restrict__ zsqw,
    float* __restrict__ rowM, float* __restrict__ rowS, int* __restrict__ idxo) {
  const int lane = threadIdx.x & 31, wave = threadIdx.x >> 5;
  const int r = lane & 15, sel = lane >> 4;
  const int h = blockIdx.y;
  const int r0 = blockIdx.x * 128 + wave * 16;

  const unsigned short* aB = zbf + ((size_t)(r0 + r) * Hh + h) * 64 + sel*8;
  v16bf a0 = ld_bf16x16(aB, 16);        // d 0..31
  v16bf a1 = ld_bf16x16(aB + 32, 16);   // d 32..63

  float zs[8];
#pragma unroll
  for (int rr = 0; rr < 8; ++rr) zs[rr] = zsqw[(size_t)(r0 + sel*8 + rr) * Hh + h];

  float mind[8], mx[8], sm[8]; int mi[8];
#pragma unroll
  for (int rr = 0; rr < 8; ++rr) { mind[rr]=3.0e38f; mx[rr]=-3.0e38f; sm[rr]=0.f; mi[rr]=0; }

  const unsigned short* cbh = cbbf + (size_t)h * Kk * 64;
  const float* csqh = csq + h * Kk;
  for (int k0 = 0; k0 < Kk; k0 += 16) {
    const unsigned short* bB = cbh + (size_t)(k0 + r) * 64 + sel*8;
    __builtin_prefetch(bB + 16*64, 0, 1);   // global_prefetch next tile
    v16bf b0 = ld_bf16x16(bB, 16);
    v16bf b1 = ld_bf16x16(bB + 32, 16);
    v8f acc = zero8();
    acc = wmma_bf16(a0, b0, acc);
    acc = wmma_bf16(a1, b1, acc);
    float cs = csqh[k0 + r];
    int ke = k0 + r;
#pragma unroll
    for (int rr = 0; rr < 8; ++rr) {
      float d = zs[rr] + cs - 2.0f * acc[rr];
      if (d < mind[rr]) { mind[rr] = d; mi[rr] = ke; }
      float neg = -d;
      if (neg > mx[rr]) { sm[rr] = sm[rr]*__expf(mx[rr]-neg) + 1.0f; mx[rr] = neg; }
      else               sm[rr] += __expf(neg - mx[rr]);
    }
  }
#pragma unroll
  for (int mask = 1; mask <= 8; mask <<= 1) {
#pragma unroll
    for (int rr = 0; rr < 8; ++rr) {
      float od = __shfl_xor(mind[rr], mask, 32);
      int   oi = __shfl_xor(mi[rr],   mask, 32);
      if (od < mind[rr] || (od == mind[rr] && oi < mi[rr])) { mind[rr]=od; mi[rr]=oi; }
      float om = __shfl_xor(mx[rr], mask, 32);
      float os = __shfl_xor(sm[rr], mask, 32);
      float nm = fmaxf(mx[rr], om);
      sm[rr] = sm[rr]*__expf(mx[rr]-nm) + os*__expf(om-nm);
      mx[rr] = nm;
    }
  }
  if (r == 0) {
#pragma unroll
    for (int rr = 0; rr < 8; ++rr) {
      size_t row = (size_t)(r0 + sel*8 + rr) * Hh + h;
      rowM[row] = mx[rr]; rowS[row] = sm[rr]; idxo[row] = mi[rr];
    }
  }
}

// ---------------------------------------------------------------------------
// k3b: recompute distances, accumulate normalized softmax probs into avg_probs
// ---------------------------------------------------------------------------
__global__ __launch_bounds__(256) void k3b_probs(
    const unsigned short* __restrict__ zbf, const unsigned short* __restrict__ cbbf,
    const float* __restrict__ csq, const float* __restrict__ zsqw,
    const float* __restrict__ rowM, const float* __restrict__ rowS,
    float* __restrict__ avgp) {
  __shared__ float ps[Kk];
  for (int i = threadIdx.x; i < Kk; i += 256) ps[i] = 0.f;
  __syncthreads();

  const int lane = threadIdx.x & 31, wave = threadIdx.x >> 5;
  const int r = lane & 15, sel = lane >> 4;
  const int h = blockIdx.y;
  const int r0 = blockIdx.x * 128 + wave * 16;

  const unsigned short* aB = zbf + ((size_t)(r0 + r) * Hh + h) * 64 + sel*8;
  v16bf a0 = ld_bf16x16(aB, 16);
  v16bf a1 = ld_bf16x16(aB + 32, 16);

  float zs[8], m8[8], is8[8];
#pragma unroll
  for (int rr = 0; rr < 8; ++rr) {
    size_t row = (size_t)(r0 + sel*8 + rr) * Hh + h;
    zs[rr] = zsqw[row]; m8[rr] = rowM[row]; is8[rr] = 1.0f / rowS[row];
  }

  const unsigned short* cbh = cbbf + (size_t)h * Kk * 64;
  const float* csqh = csq + h * Kk;
  for (int k0 = 0; k0 < Kk; k0 += 16) {
    const unsigned short* bB = cbh + (size_t)(k0 + r) * 64 + sel*8;
    __builtin_prefetch(bB + 16*64, 0, 1);
    v16bf b0 = ld_bf16x16(bB, 16);
    v16bf b1 = ld_bf16x16(bB + 32, 16);
    v8f acc = zero8();
    acc = wmma_bf16(a0, b0, acc);
    acc = wmma_bf16(a1, b1, acc);
    float cs = csqh[k0 + r];
    float psum = 0.f;
#pragma unroll
    for (int rr = 0; rr < 8; ++rr) {
      float d = zs[rr] + cs - 2.0f * acc[rr];
      psum += __expf(-d - m8[rr]) * is8[rr];    // prob, per row, summed over rows
    }
    psum += __shfl_xor(psum, 16, 32);           // fold two row-halves per entry
    if (sel == 0) atomicAdd(&ps[k0 + r], psum);
  }
  __syncthreads();
  for (int i = threadIdx.x; i < Kk; i += 256) atomicAdd(&avgp[h*Kk + i], ps[i]);
}

// ---------------------------------------------------------------------------
// k4: z_proj & z_q_dec = (z_heads | codebook[idx]) @ dec_w[h] + dec_b
//     writes z_q_st (== z_q_dec forward value); accumulates commit loss
// ---------------------------------------------------------------------------
__global__ __launch_bounds__(256) void k4_dec(
    const unsigned short* __restrict__ zbf, const unsigned short* __restrict__ cbbf,
    const unsigned short* __restrict__ dwt, const float* __restrict__ decb,
    const int* __restrict__ idxv, float* __restrict__ outq, float* __restrict__ lossAcc) {
  const int lane = threadIdx.x & 31, wave = threadIdx.x >> 5;
  const int r = lane & 15, sel = lane >> 4;
  const int h = blockIdx.y;
  const int r0 = blockIdx.x * 128 + wave * 16;

  const unsigned short* aB = zbf + ((size_t)(r0 + r) * Hh + h) * 64 + sel*8;
  v16bf ap0 = ld_bf16x16(aB, 16), ap1 = ld_bf16x16(aB + 32, 16);
  int myi = idxv[(size_t)(r0 + r) * Hh + h];
  const unsigned short* aQ = cbbf + ((size_t)h * Kk + myi) * 64 + sel*8;
  v16bf aq0 = ld_bf16x16(aQ, 16), aq1 = ld_bf16x16(aQ + 32, 16);

  float loss = 0.f;
  for (int n0 = 0; n0 < 128; n0 += 16) {
    const unsigned short* bB = dwt + ((size_t)h*128 + n0 + r) * 64 + sel*8;
    v16bf b0 = ld_bf16x16(bB, 16), b1 = ld_bf16x16(bB + 32, 16);
    v8f aP = zero8(), aQc = zero8();
    aP  = wmma_bf16(ap0, b0, aP);  aP  = wmma_bf16(ap1, b1, aP);
    aQc = wmma_bf16(aq0, b0, aQc); aQc = wmma_bf16(aq1, b1, aQc);
    float bias = decb[h*128 + n0 + r];
    int col = h*128 + n0 + r;
#pragma unroll
    for (int rr = 0; rr < 8; ++rr) {
      int row = r0 + sel*8 + rr;
      float qv = aQc[rr] + bias;
      float pv = aP[rr] + bias;
      outq[(size_t)row * HIDN + col] = qv;   // z_q_st forward value == z_q_dec
      float df = pv - qv;
      loss += df * df;
    }
  }
#pragma unroll
  for (int mask = 1; mask <= 16; mask <<= 1) loss += __shfl_xor(loss, mask, 32);
  if (lane == 0) atomicAdd(lossAcc, loss);
}

// ---------------------------------------------------------------------------
// Histogram + ortho (sparse co-occurrence sweep) + finalize scalars
// ---------------------------------------------------------------------------
__global__ void k_hist(const int* __restrict__ idxv, float* __restrict__ pcnt) {
  int i = blockIdx.x * blockDim.x + threadIdx.x;
  if (i < Bb*Hh) atomicAdd(&pcnt[(i & 7) * Kk + idxv[i]], 1.0f);
}

__global__ void k_oscatter(const int* __restrict__ idxv, float* __restrict__ co,
                           int i, int j) {
  int b = blockIdx.x * blockDim.x + threadIdx.x;
  if (b < Bb) atomicAdd(&co[(size_t)idxv[b*Hh+i] * Kk + idxv[b*Hh+j]], 1.0f);
}

__global__ void k_ocollect(const int* __restrict__ idxv, float* __restrict__ co,
                           const float* __restrict__ pcnt, float* __restrict__ scal,
                           int i, int j) {
  int b = blockIdx.x * blockDim.x + threadIdx.x;
  if (b < Bb) {
    int ki = idxv[b*Hh+i], kj = idxv[b*Hh+j];
    float old = atomicExch(&co[(size_t)ki * Kk + kj], 0.0f);  // claim + reset
    if (old != 0.0f) {
      float c = old * (1.0f / 8192.0f);
      float e = pcnt[i*Kk+ki] * pcnt[j*Kk+kj] * (1.0f / (8192.0f * 8192.0f));
      atomicAdd(&scal[1], fabsf(c - e) - e);
    }
  }
}

__global__ void k_fin(const float* __restrict__ avgp, const float* __restrict__ scal,
                      float* __restrict__ outp) {
  __shared__ float red[256];
  float e = 0.f;
  for (int i = threadIdx.x; i < Hh*Kk; i += 256) {
    float a = avgp[i] * (1.0f / 8192.0f);
    e -= a * logf(a + 1e-8f);
  }
  red[threadIdx.x] = e;
  __syncthreads();
  for (int s = 128; s > 0; s >>= 1) {
    if (threadIdx.x < s) red[threadIdx.x] += red[threadIdx.x + s];
    __syncthreads();
  }
  if (threadIdx.x == 0) {
    float entropy = red[0] / (float)Hh;
    float el = 1.0f - entropy / logf(2048.0f);
    float commit = scal[0] / (8192.0f * 8.0f * 128.0f);
    // mean|co - e| = (sum_touched(|co-e|-e) + 1)/K^2 per pair; 28 pairs
    float orth = (scal[1] + 28.0f) / (28.0f * 2048.0f * 2048.0f);
    outp[8454144] = commit;   // commit
    outp[8454145] = commit;   // codebook_loss (numerically identical fwd value)
    outp[8454146] = el;       // entropy_loss
    outp[8454147] = orth;     // ortho
  }
}

// ---------------------------------------------------------------------------
// Host launcher
// ---------------------------------------------------------------------------
extern "C" void kernel_launch(void* const* d_in, const int* in_sizes, int n_in,
                              void* d_out, int out_size, void* d_ws, size_t ws_size,
                              hipStream_t stream) {
  (void)in_sizes; (void)n_in; (void)out_size; (void)ws_size;
  const float* z    = (const float*)d_in[0];
  const float* w1   = (const float*)d_in[1];
  const float* b1   = (const float*)d_in[2];
  const float* w2   = (const float*)d_in[3];
  const float* b2   = (const float*)d_in[4];
  const float* lng  = (const float*)d_in[5];
  const float* lnb  = (const float*)d_in[6];
  const float* cb   = (const float*)d_in[7];
  const float* decw = (const float*)d_in[8];
  const float* decb = (const float*)d_in[9];

  char* ws = (char*)d_ws;
  unsigned short* h1bf = (unsigned short*)(ws + 0);        // 16777216 B
  unsigned short* zbf  = (unsigned short*)(ws + 16777216); // 16777216 B
  unsigned short* zhbf = (unsigned short*)(ws + 33554432); //  8388608 B
  unsigned short* w1t  = (unsigned short*)(ws + 41943040); //  2097152 B
  unsigned short* w2t  = (unsigned short*)(ws + 44040192); //   131072 B
  unsigned short* dwt  = (unsigned short*)(ws + 44171264); //   131072 B
  unsigned short* cbbf = (unsigned short*)(ws + 44302336); //  2097152 B
  float*          csq  = (float*)(ws + 46399488);          //    65536 B
  float*          rowM = (float*)(ws + 46465024);          //   262144 B
  float*          rowS = (float*)(ws + 46727168);          //   262144 B
  float*          zsq  = (float*)(ws + 46989312);          //   262144 B
  float*          avgp = (float*)(ws + 47251456);          //    65536 B
  float*          pcnt = (float*)(ws + 47316992);          //    65536 B
  float*          scal = (float*)(ws + 47382528);          //      256 B
  float*          co   = (float*)(ws + 47382784);          // 16777216 B

  float* outp  = (float*)d_out;
  float* outq  = outp;                       // z_q_st : 8388608 floats
  int*   idxo  = (int*)(outp + 8388608);     // idx    :   65536 int32
  float* outzh = outp + 8454148;             // z_heads: 4194304 floats

  k_prep<<<4096, 256, 0, stream>>>(z, w1, w2, decw, cb, zbf, w1t, w2t, dwt, cbbf,
                                   co, avgp, pcnt, scal);
  k_csq<<<64, 256, 0, stream>>>(cb, csq);
  k1_gemm1_gelu<<<dim3(8, 64), 256, 0, stream>>>(zbf, w1t, b1, h1bf);
  k2_gemm2_ln<<<dim3(8, 64), 256, 0, stream>>>(h1bf, w2t, b2, lng, lnb,
                                               zhbf, zsq, outzh);
  k3a_dist<<<dim3(64, 8), 256, 0, stream>>>(zhbf, cbbf, csq, zsq, rowM, rowS, idxo);
  k3b_probs<<<dim3(64, 8), 256, 0, stream>>>(zhbf, cbbf, csq, zsq, rowM, rowS, avgp);
  k4_dec<<<dim3(64, 8), 256, 0, stream>>>(zhbf, cbbf, dwt, decb, idxo, outq, scal);
  k_hist<<<256, 256, 0, stream>>>(idxo, pcnt);
  for (int i = 0; i < Hh; ++i)
    for (int j = i + 1; j < Hh; ++j) {
      k_oscatter<<<32, 256, 0, stream>>>(idxo, co, i, j);
      k_ocollect<<<32, 256, 0, stream>>>(idxo, co, pcnt, scal, i, j);
    }
  k_fin<<<1, 256, 0, stream>>>(avgp, scal, outp);
}
